// TransformerDecoderLayerQaN_8581344657657
// MI455X (gfx1250) — compile-verified, matrix operand order checked by reading, not run on hardware
//
#include <hip/hip_runtime.h>
#include <hip/hip_bf16.h>
#include <math.h>
#include <stdint.h>

typedef __attribute__((ext_vector_type(16))) _Float16 v16h;
typedef __attribute__((ext_vector_type(8)))  _Float16 v8h;
typedef __attribute__((ext_vector_type(8)))  float    v8f;

#define T_LEN 128
#define B_TOT 480
#define D_MOD 256
#define NHEADS 8
#define HDIM 32
#define NQ_ 10
#define NP_ 3
#define B0_ 16
#define WSZ 16
#define NW 8
#define DFF_ 1024
#define NTOK (T_LEN * B_TOT)      /* 61440 */
#define MMB  (T_LEN * B0_ * NQ_)  /* 20480 */

// ---------------------------------------------------------------------------
// CDNA5 async global->LDS copy (GLOBAL_LOAD_ASYNC_TO_LDS_B128, ASYNCcnt).
// VDST VGPR carries the LDS byte address: low 32 bits of a flat LDS pointer
// are exactly the LDS offset (ISA 10.2 aperture rules).
// ---------------------------------------------------------------------------
__device__ __forceinline__ void async_ld_b128(void* lds, const void* gaddr) {
  asm volatile("global_load_async_to_lds_b128 %0, %1, off"
               :: "v"((unsigned)(uintptr_t)lds), "v"(gaddr)
               : "memory");
}
__device__ __forceinline__ void wait_async() {
  asm volatile("s_wait_asynccnt 0x0" ::: "memory");
}

// ---------------------------------------------------------------------------
// WMMA fragment loader: row-major [16][ld] tile of f16 in LDS.
// A: row = M (lane&15). B passed as B^T stored [N][K] row-major, row = N.
// Lane half (lane>>4) selects k-offsets {0..7,16..23} / {8..15,24..31}:
// two contiguous 8-half groups -> two 16B LDS loads.
// ---------------------------------------------------------------------------
__device__ __forceinline__ v16h frag_ld(const _Float16* tile, int ld, int lane) {
  const int r  = lane & 15;
  const int hs = (lane >> 4) << 3;
  const _Float16* p = tile + r * ld + hs;
  v8h lo = *(const v8h*)p;
  v8h hi = *(const v8h*)(p + 16);
  v16h a;
#pragma unroll
  for (int i = 0; i < 8; ++i) { a[i] = lo[i]; a[i + 8] = hi[i]; }
  return a;
}

// ---------------------------------------------------------------------------
// Generic WMMA GEMM: C[M,N] = A[M,K] (f16) * BT[N,K]^T (f16) + bias, optional
// ReLU, optional A-row gather map and C-row scatter map. Block = 128 threads
// (4 waves). Tile = 32 rows x 64 cols: each wave owns one 16-col n-tile and
// two 16-row m-subtiles (2 accumulators -> 4 WMMAs per staged K-chunk).
// ---------------------------------------------------------------------------
template <bool RELU, bool C16>
__global__ __launch_bounds__(128) void gemm_wmma(
    const _Float16* __restrict__ A, const int* __restrict__ amap,
    const _Float16* __restrict__ BT, const float* __restrict__ bias,
    void* __restrict__ C, const int* __restrict__ cmap, int K, int ldc) {
  __shared__ __attribute__((aligned(16))) _Float16 As[32][64];
  __shared__ __attribute__((aligned(16))) _Float16 Bs[4][16][64];

  const int tid  = threadIdx.x;
  const int wave = tid >> 5;
  const int lane = tid & 31;
  const int m0 = blockIdx.x * 32;
  const int n0 = blockIdx.y * 64;

  v8f acc0 = {}, acc1 = {};
  for (int kb = 0; kb < K; kb += 64) {
    // stage A chunk: 32x64 halves, two b128 async copies per thread
#pragma unroll
    for (int it = 0; it < 2; ++it) {
      const int idx = (it * 128 + tid) * 8;
      const int r = idx >> 6, c = idx & 63;
      const int arow = amap ? amap[m0 + r] : (m0 + r);
      async_ld_b128(&As[r][c], A + (long long)arow * K + kb + c);
    }
    // each wave stages its own 16x64 B^T chunk
#pragma unroll
    for (int it = 0; it < 4; ++it) {
      const int idx = (it * 32 + lane) * 8;
      const int r = idx >> 6, c = idx & 63;
      const int nrow = n0 + wave * 16 + r;
      async_ld_b128(&Bs[wave][r][c], BT + (long long)nrow * K + kb + c);
    }
    if (kb + 64 < K)
      __builtin_prefetch((const void*)(BT + (long long)(n0 + wave * 16) * K + kb + 64), 0, 1);
    wait_async();
    __syncthreads();
#pragma unroll
    for (int k0 = 0; k0 < 64; k0 += 32) {
      v16h b  = frag_ld(&Bs[wave][0][0] + k0, 64, lane);
      v16h a0 = frag_ld(&As[0][0]  + k0, 64, lane);
      v16h a1 = frag_ld(&As[16][0] + k0, 64, lane);
      acc0 = __builtin_amdgcn_wmma_f32_16x16x32_f16(false, a0, false, b, (short)0,
                                                    acc0, false, false);
      acc1 = __builtin_amdgcn_wmma_f32_16x16x32_f16(false, a1, false, b, (short)0,
                                                    acc1, false, false);
    }
    __syncthreads();
  }

  const int nl = lane & 15, mb = (lane >> 4) << 3;
  const int n = n0 + wave * 16 + nl;
  const float bv = bias ? bias[n] : 0.f;
#pragma unroll
  for (int v = 0; v < 8; ++v) {
    const int mr0 = m0 + mb + v;
    const int mr1 = m0 + 16 + mb + v;
    const int cr0 = cmap ? cmap[mr0] : mr0;
    const int cr1 = cmap ? cmap[mr1] : mr1;
    float v0 = acc0[v] + bv, v1 = acc1[v] + bv;
    if (RELU) { v0 = fmaxf(v0, 0.f); v1 = fmaxf(v1, 0.f); }
    if (C16) {
      ((_Float16*)C)[(long long)cr0 * ldc + n] = (_Float16)v0;
      ((_Float16*)C)[(long long)cr1 * ldc + n] = (_Float16)v1;
    } else {
      ((float*)C)[(long long)cr0 * ldc + n] = v0;
      ((float*)C)[(long long)cr1 * ldc + n] = v1;
    }
  }
}

// ---------------------------------------------------------------------------
// qa_block: rotary + local attention (window 16, look +-1), one wave per
// (batch bb, window w). Q is the normalized learned query broadcast over t.
// ---------------------------------------------------------------------------
__global__ __launch_bounds__(32) void qa_local_attn(
    const float* __restrict__ x, const float* __restrict__ queries,
    float* __restrict__ y) {
  __shared__ __attribute__((aligned(16))) _Float16 Qs[16][D_MOD];
  __shared__ __attribute__((aligned(16))) _Float16 Ks[48][D_MOD];
  __shared__ __attribute__((aligned(16))) _Float16 Vt[D_MOD][64];
  __shared__ __attribute__((aligned(16))) _Float16 Ps[16][64];
  __shared__ float sims[16][48];
  __shared__ float hn[NHEADS];

  const int bb = blockIdx.x;  // ((b0*NP+np)*NQ+nq)
  const int w  = blockIdx.y;
  const int lane = threadIdx.x;

  const int nq = bb % NQ_;
  const int np = (bb / NQ_) % NP_;
  const int b0 = bb / (NQ_ * NP_);
  const int bcol = (b0 * NQ_ + nq) * NP_ + np;

  const float* qrow = queries + nq * D_MOD;

  if (lane < NHEADS) {
    float s = 0.f;
    for (int d = 0; d < HDIM; ++d) { float v = qrow[lane * HDIM + d]; s += v * v; }
    hn[lane] = sqrtf(s);
  }
  __syncthreads();

  const float kLog   = 0.07195578459f;  // ln(10000)/128
  const float qscale = 0.1767766953f;   // 1/sqrt(32)

  // rotary-embedded Q tile (16 x 256)
  for (int r = 0; r < 16; ++r) {
    const int t = w * WSZ + r;
    for (int d = lane; d < D_MOD; d += 32) {
      const int dp = d ^ 128;
      const float qv = qrow[d]  / (hn[d  >> 5] + 1e-6f) * qscale;
      const float qp = qrow[dp] / (hn[dp >> 5] + 1e-6f) * qscale;
      const float rh = (d < 128) ? -qp : qp;
      const float ang = (float)t * __expf(-(float)(d & 127) * kLog);
      Qs[r][d] = (_Float16)(qv * cosf(ang) + rh * sinf(ang));
    }
  }
  // rotary K (48 keys) + V^T (un-rotated)
  for (int j = 0; j < 48; ++j) {
    const int tp = (w - 1) * WSZ + j;
    const bool valid = (tp >= 0) && (tp < T_LEN);
    const float* xr = x + ((long long)(valid ? tp : 0) * B_TOT + bcol) * D_MOD;
    for (int d = lane; d < D_MOD; d += 32) {
      float xv = 0.f, xq = 0.f;
      if (valid) { xv = xr[d]; xq = xr[d ^ 128]; }
      const float rh = (d < 128) ? -xq : xq;
      const float ang = (float)tp * __expf(-(float)(d & 127) * kLog);
      Ks[j][d] = (_Float16)(valid ? (xv * cosf(ang) + rh * sinf(ang)) : 0.f);
      Vt[d][j] = (_Float16)xv;
    }
  }
  for (int d = lane; d < D_MOD; d += 32)
    for (int j = 48; j < 64; ++j) Vt[d][j] = (_Float16)0.f;
  __syncthreads();

  // sim = Q K^T / 16, with look-around padding masked to -inf
  for (int j0 = 0; j0 < 48; j0 += 16) {
    v8f acc = {};
#pragma unroll
    for (int k0 = 0; k0 < D_MOD; k0 += 32) {
      v16h a = frag_ld(&Qs[0][0]  + k0, D_MOD, lane);
      v16h b = frag_ld(&Ks[j0][0] + k0, D_MOD, lane);
      acc = __builtin_amdgcn_wmma_f32_16x16x32_f16(false, a, false, b, (short)0,
                                                   acc, false, false);
    }
    const int nl = lane & 15, mb = (lane >> 4) << 3;
    const int j = j0 + nl;
    const int tp = (w - 1) * WSZ + j;
    const bool valid = (tp >= 0) && (tp < T_LEN);
#pragma unroll
    for (int v = 0; v < 8; ++v)
      sims[mb + v][j] = valid ? (acc[v] * (1.f / 16.f)) : -1e30f;
  }
  __syncthreads();

  if (lane < 16) {
    float mx = -1e30f;
    for (int j = 0; j < 48; ++j) mx = fmaxf(mx, sims[lane][j]);
    float ssum = 0.f;
    for (int j = 0; j < 48; ++j) ssum += __expf(sims[lane][j] - mx);
    const float inv = 1.f / ssum;
    for (int j = 0; j < 48; ++j)
      Ps[lane][j] = (_Float16)(__expf(sims[lane][j] - mx) * inv);
    for (int j = 48; j < 64; ++j) Ps[lane][j] = (_Float16)0.f;
  }
  __syncthreads();

  // out = P V  (K padded to 64 with zeros)
  for (int d0 = 0; d0 < D_MOD; d0 += 16) {
    v8f acc = {};
#pragma unroll
    for (int k0 = 0; k0 < 64; k0 += 32) {
      v16h a = frag_ld(&Ps[0][0]  + k0, 64, lane);
      v16h b = frag_ld(&Vt[d0][0] + k0, 64, lane);
      acc = __builtin_amdgcn_wmma_f32_16x16x32_f16(false, a, false, b, (short)0,
                                                   acc, false, false);
    }
    const int nl = lane & 15, mb = (lane >> 4) << 3;
#pragma unroll
    for (int v = 0; v < 8; ++v) {
      const int t = w * WSZ + mb + v;
      y[((long long)t * B_TOT + bcol) * D_MOD + d0 + nl] = acc[v];
    }
  }
}

// ---------------------------------------------------------------------------
// Cross attention: one wave per (16-row q tile, head, batch column).
// Q/K staged with async global->LDS b128 copies; V staged transposed.
// ---------------------------------------------------------------------------
__global__ __launch_bounds__(32) void mha_attn(
    const _Float16* __restrict__ Qc, const _Float16* __restrict__ Kc,
    const _Float16* __restrict__ Vc, _Float16* __restrict__ Oc) {
  __shared__ __attribute__((aligned(16))) _Float16 Qs[16][HDIM];
  __shared__ __attribute__((aligned(16))) _Float16 Ks[T_LEN][HDIM];
  __shared__ __attribute__((aligned(16))) _Float16 Vt[HDIM][T_LEN];
  __shared__ __attribute__((aligned(16))) _Float16 Ps[16][T_LEN];
  __shared__ float sims[16][T_LEN];

  const int lt = blockIdx.x;
  const int h  = blockIdx.y;
  const int bcol = blockIdx.z;
  const int lane = threadIdx.x;

#pragma unroll
  for (int it = 0; it < 2; ++it) {  // Qs: 16x32 halves, b128 chunks
    const int idx = (it * 32 + lane) * 8;
    const int r = idx >> 5, c = idx & 31;
    const int t = lt * 16 + r;
    async_ld_b128(&Qs[r][c],
                  Qc + ((long long)t * B_TOT + bcol) * D_MOD + h * HDIM + c);
  }
#pragma unroll
  for (int it = 0; it < 16; ++it) {  // Ks: 128x32 halves, b128 chunks
    const int idx = (it * 32 + lane) * 8;
    const int s = idx >> 5, c = idx & 31;
    async_ld_b128(&Ks[s][c],
                  Kc + ((long long)s * B_TOT + bcol) * D_MOD + h * HDIM + c);
  }
  for (int idx = lane; idx < T_LEN * HDIM; idx += 32) {  // V transposed
    const int s = idx >> 5, d = idx & 31;
    Vt[d][s] = Vc[((long long)s * B_TOT + bcol) * D_MOD + h * HDIM + d];
  }
  wait_async();
  __syncthreads();

  const float scale = 0.1767766953f;  // 1/sqrt(32)
  for (int s0 = 0; s0 < T_LEN; s0 += 16) {
    v8f acc = {};
    v16h a = frag_ld(&Qs[0][0], HDIM, lane);
    v16h b = frag_ld(&Ks[s0][0], HDIM, lane);
    acc = __builtin_amdgcn_wmma_f32_16x16x32_f16(false, a, false, b, (short)0,
                                                 acc, false, false);
    const int nl = lane & 15, mb = (lane >> 4) << 3;
#pragma unroll
    for (int v = 0; v < 8; ++v) sims[mb + v][s0 + nl] = acc[v] * scale;
  }
  __syncthreads();

  if (lane < 16) {
    float mx = -1e30f;
    for (int s = 0; s < T_LEN; ++s) mx = fmaxf(mx, sims[lane][s]);
    float ssum = 0.f;
    for (int s = 0; s < T_LEN; ++s) ssum += __expf(sims[lane][s] - mx);
    const float inv = 1.f / ssum;
    for (int s = 0; s < T_LEN; ++s)
      Ps[lane][s] = (_Float16)(__expf(sims[lane][s] - mx) * inv);
  }
  __syncthreads();

  for (int d0 = 0; d0 < HDIM; d0 += 16) {
    v8f acc = {};
#pragma unroll
    for (int k0 = 0; k0 < T_LEN; k0 += 32) {
      v16h a = frag_ld(&Ps[0][0]  + k0, T_LEN, lane);
      v16h b = frag_ld(&Vt[d0][0] + k0, T_LEN, lane);
      acc = __builtin_amdgcn_wmma_f32_16x16x32_f16(false, a, false, b, (short)0,
                                                   acc, false, false);
    }
    const int nl = lane & 15, mb = (lane >> 4) << 3;
#pragma unroll
    for (int v = 0; v < 8; ++v) {
      const int t = lt * 16 + mb + v;
      Oc[((long long)t * B_TOT + bcol) * D_MOD + h * HDIM + d0 + nl] = acc[v];
    }
  }
}

// ---------------------------------------------------------------------------
// mm_block inner attention: seq len 3, one thread per (batch m, head h).
// qkv layout: rows = p*MMB+m (mm-token order), cols [q(256) | k(256) | v(256)].
// ---------------------------------------------------------------------------
__global__ void mm_attn(const _Float16* __restrict__ qkv,
                        _Float16* __restrict__ ao) {
  const int i = blockIdx.x * blockDim.x + threadIdx.x;
  if (i >= MMB * NHEADS) return;
  const int h = i & (NHEADS - 1);
  const int m = i >> 3;
  const float scale = 0.1767766953f;

  float s[NP_][NP_];
#pragma unroll
  for (int p = 0; p < NP_; ++p)
#pragma unroll
    for (int q = 0; q < NP_; ++q) {
      const _Float16* qp = qkv + (long long)(p * MMB + m) * 768 + h * HDIM;
      const _Float16* kp = qkv + (long long)(q * MMB + m) * 768 + 256 + h * HDIM;
      float acc = 0.f;
      for (int d = 0; d < HDIM; ++d) acc += (float)qp[d] * (float)kp[d];
      s[p][q] = acc * scale;
    }
  const _Float16* v0 = qkv + (long long)(0 * MMB + m) * 768 + 512 + h * HDIM;
  const _Float16* v1 = qkv + (long long)(1 * MMB + m) * 768 + 512 + h * HDIM;
  const _Float16* v2 = qkv + (long long)(2 * MMB + m) * 768 + 512 + h * HDIM;
#pragma unroll
  for (int p = 0; p < NP_; ++p) {
    const float mx = fmaxf(s[p][0], fmaxf(s[p][1], s[p][2]));
    const float e0 = __expf(s[p][0] - mx), e1 = __expf(s[p][1] - mx),
                e2 = __expf(s[p][2] - mx);
    const float inv = 1.f / (e0 + e1 + e2);
    const float a0 = e0 * inv, a1 = e1 * inv, a2 = e2 * inv;
    _Float16* op = ao + (long long)(p * MMB + m) * D_MOD + h * HDIM;
    for (int d = 0; d < HDIM; ++d)
      op[d] = (_Float16)(a0 * (float)v0[d] + a1 * (float)v1[d] + a2 * (float)v2[d]);
  }
}

// ---------------------------------------------------------------------------
// Fused residual + LayerNorm (one wave per token, D=256). In-place safe.
// ---------------------------------------------------------------------------
__global__ __launch_bounds__(128) void add_ln(
    const float* __restrict__ x, const float* __restrict__ y,
    const float* __restrict__ yscale_ptr, const float* __restrict__ g,
    const float* __restrict__ bvec, float* __restrict__ o32,
    _Float16* __restrict__ o16) {
  const int tok = blockIdx.x * 4 + (threadIdx.x >> 5);
  const int lane = threadIdx.x & 31;
  const float ys = yscale_ptr ? yscale_ptr[0] : 1.0f;
  const float* xp = x + (long long)tok * D_MOD;
  const float* yp = y + (long long)tok * D_MOD;

  float v[8];
  float s = 0.f;
#pragma unroll
  for (int i = 0; i < 8; ++i) {
    const int d = lane + i * 32;
    v[i] = xp[d] + ys * yp[d];
    s += v[i];
  }
#pragma unroll
  for (int off = 16; off > 0; off >>= 1) s += __shfl_xor(s, off, 32);
  const float mu = s * (1.f / 256.f);
  float var = 0.f;
#pragma unroll
  for (int i = 0; i < 8; ++i) { const float dd = v[i] - mu; var += dd * dd; }
#pragma unroll
  for (int off = 16; off > 0; off >>= 1) var += __shfl_xor(var, off, 32);
  const float inv = rsqrtf(var * (1.f / 256.f) + 1e-5f);
#pragma unroll
  for (int i = 0; i < 8; ++i) {
    const int d = lane + i * 32;
    const float o = (v[i] - mu) * inv * g[d] + bvec[d];
    o32[(long long)tok * D_MOD + d] = o;
    if (o16) o16[(long long)tok * D_MOD + d] = (_Float16)o;
  }
}

__global__ void cvt_f16(const float* __restrict__ s, _Float16* __restrict__ d,
                        int n) {
  const int i = blockIdx.x * blockDim.x + threadIdx.x;
  if (i < n) d[i] = (_Float16)s[i];
}

// mm-token i = p*MMB + (t*B0+b0)*NQ+nq  ->  x-token t*B + (b0*NQ+nq)*NP + p
__global__ void mk_map(int* __restrict__ map) {
  const int i = blockIdx.x * blockDim.x + threadIdx.x;
  if (i >= NP_ * MMB) return;
  const int p = i / MMB;
  const int m = i % MMB;
  const int nq = m % NQ_;
  const int mb = m / NQ_;
  const int b0 = mb % B0_;
  const int t  = mb / B0_;
  map[i] = t * B_TOT + (b0 * NQ_ + nq) * NP_ + p;
}

// ---------------------------------------------------------------------------
extern "C" void kernel_launch(void* const* d_in, const int* in_sizes, int n_in,
                              void* d_out, int out_size, void* d_ws,
                              size_t ws_size, hipStream_t stream) {
  (void)in_sizes; (void)n_in; (void)out_size; (void)ws_size;
  const float* tgt       = (const float*)d_in[0];
  const float* memory    = (const float*)d_in[1];
  const float* queries   = (const float*)d_in[2];
  const float* mha_in_w  = (const float*)d_in[3];
  const float* mha_in_b  = (const float*)d_in[4];
  const float* mha_out_w = (const float*)d_in[5];
  const float* mha_out_b = (const float*)d_in[6];
  const float* mm_in_w   = (const float*)d_in[7];
  const float* mm_in_b   = (const float*)d_in[8];
  const float* mm_out_w  = (const float*)d_in[9];
  const float* mm_out_b  = (const float*)d_in[10];
  const float* lin1_w    = (const float*)d_in[11];
  const float* lin1_b    = (const float*)d_in[12];
  const float* lin2_w    = (const float*)d_in[13];
  const float* lin2_b    = (const float*)d_in[14];
  const float* n1_g = (const float*)d_in[15];
  const float* n1_b = (const float*)d_in[16];
  const float* n2_g = (const float*)d_in[17];
  const float* n2_b = (const float*)d_in[18];
  const float* n3_g = (const float*)d_in[19];
  const float* n3_b = (const float*)d_in[20];
  const float* n4_g = (const float*)d_in[21];
  const float* n4_b = (const float*)d_in[22];
  const float* glvl = (const float*)d_in[23];

  char* ws = (char*)d_ws;
  size_t off = 0;
  auto alloc = [&](size_t bytes) -> char* {
    char* p = ws + off;
    off = (off + bytes + 255) & ~(size_t)255;
    return p;
  };

  _Float16* w_mha16  = (_Float16*)alloc((size_t)768 * 256 * 2);
  _Float16* w_mhao16 = (_Float16*)alloc((size_t)256 * 256 * 2);
  _Float16* w_mmin16 = (_Float16*)alloc((size_t)768 * 256 * 2);
  _Float16* w_mmo16  = (_Float16*)alloc((size_t)256 * 256 * 2);
  _Float16* w_l1_16  = (_Float16*)alloc((size_t)1024 * 256 * 2);
  _Float16* w_l2_16  = (_Float16*)alloc((size_t)1024 * 256 * 2);
  _Float16* mem16    = (_Float16*)alloc((size_t)NTOK * 256 * 2);
  int*      map_mm   = (int*)alloc((size_t)NTOK * 4);
  float*    ybuf     = (float*)alloc((size_t)NTOK * 256 * 4);
  float*    x32      = (float*)alloc((size_t)NTOK * 256 * 4);
  _Float16* x16      = (_Float16*)alloc((size_t)NTOK * 256 * 2);
  _Float16* big      = (_Float16*)alloc((size_t)NTOK * 1024 * 2);  // qkv / QKV / ffn-h
  _Float16* small16  = (_Float16*)alloc((size_t)NTOK * 256 * 2);   // ao / Oc

  const dim3 c256(256);
  cvt_f16<<<(768 * 256 + 255) / 256, c256, 0, stream>>>(mha_in_w, w_mha16, 768 * 256);
  cvt_f16<<<(256 * 256 + 255) / 256, c256, 0, stream>>>(mha_out_w, w_mhao16, 256 * 256);
  cvt_f16<<<(768 * 256 + 255) / 256, c256, 0, stream>>>(mm_in_w, w_mmin16, 768 * 256);
  cvt_f16<<<(256 * 256 + 255) / 256, c256, 0, stream>>>(mm_out_w, w_mmo16, 256 * 256);
  cvt_f16<<<(1024 * 256 + 255) / 256, c256, 0, stream>>>(lin1_w, w_l1_16, 1024 * 256);
  cvt_f16<<<(1024 * 256 + 255) / 256, c256, 0, stream>>>(lin2_w, w_l2_16, 1024 * 256);
  cvt_f16<<<(NTOK * 256 + 255) / 256, c256, 0, stream>>>(memory, mem16, NTOK * 256);
  mk_map<<<(NTOK + 255) / 256, c256, 0, stream>>>(map_mm);

  // ---- stage 1: qa_block + LN(n1) ----
  qa_local_attn<<<dim3(B_TOT, NW), 32, 0, stream>>>(tgt, queries, ybuf);
  add_ln<<<NTOK / 4, 128, 0, stream>>>(tgt, ybuf, nullptr, n1_g, n1_b, x32, x16);

  // ---- stage 2: mm_block + LN(n4), scaled by global_level ----
  gemm_wmma<false, true><<<dim3(NTOK / 32, 768 / 64), 128, 0, stream>>>(
      x16, map_mm, w_mmin16, mm_in_b, big, nullptr, 256, 768);
  mm_attn<<<(MMB * NHEADS + 255) / 256, c256, 0, stream>>>(big, small16);
  gemm_wmma<false, false><<<dim3(NTOK / 32, 256 / 64), 128, 0, stream>>>(
      small16, nullptr, w_mmo16, mm_out_b, ybuf, map_mm, 256, 256);
  add_ln<<<NTOK / 4, 128, 0, stream>>>(x32, ybuf, glvl, n4_g, n4_b, x32, x16);

  // ---- stage 3: cross attention + LN(n2) ----
  _Float16* Qc = big;
  _Float16* Kc = big + (size_t)NTOK * 256;
  _Float16* Vc = big + (size_t)2 * NTOK * 256;
  gemm_wmma<false, true><<<dim3(NTOK / 32, 256 / 64), 128, 0, stream>>>(
      x16, nullptr, w_mha16, mha_in_b, Qc, nullptr, 256, 256);
  gemm_wmma<false, true><<<dim3(NTOK / 32, 256 / 64), 128, 0, stream>>>(
      mem16, nullptr, w_mha16 + 256 * 256, mha_in_b + 256, Kc, nullptr, 256, 256);
  gemm_wmma<false, true><<<dim3(NTOK / 32, 256 / 64), 128, 0, stream>>>(
      mem16, nullptr, w_mha16 + 512 * 256, mha_in_b + 512, Vc, nullptr, 256, 256);
  mha_attn<<<dim3(T_LEN / 16, NHEADS, B_TOT), 32, 0, stream>>>(Qc, Kc, Vc, small16);
  gemm_wmma<false, false><<<dim3(NTOK / 32, 256 / 64), 128, 0, stream>>>(
      small16, nullptr, w_mhao16, mha_out_b, ybuf, nullptr, 256, 256);
  add_ln<<<NTOK / 4, 128, 0, stream>>>(x32, ybuf, nullptr, n2_g, n2_b, x32, x16);

  // ---- stage 4: FFN + LN(n3) -> d_out ----
  gemm_wmma<true, true><<<dim3(NTOK / 32, 1024 / 64), 128, 0, stream>>>(
      x16, nullptr, w_l1_16, lin1_b, big, nullptr, 256, 1024);
  gemm_wmma<false, false><<<dim3(NTOK / 32, 256 / 64), 128, 0, stream>>>(
      big, nullptr, w_l2_16, lin2_b, ybuf, nullptr, 1024, 256);
  add_ln<<<NTOK / 4, 128, 0, stream>>>(x32, ybuf, nullptr, n3_g, n3_b,
                                       (float*)d_out, nullptr);
}
